// MoELayer_9981503996001
// MI455X (gfx1250) — matrix operand order repaired and checked by reading, not compiled, not run on hardware
//
#include <hip/hip_runtime.h>
#include <math.h>

// Problem constants (reference: B=4,S=2048,H=2048,E=4,K=2,FF=4H)
#define T_TOK 8192
#define H_DIM 2048
#define FF_DIM 8192
#define N_EXP 4
#define TOP_K 2

typedef __attribute__((ext_vector_type(16))) __bf16       v16bf;
typedef __attribute__((ext_vector_type(8)))  float        v8f;
typedef __attribute__((ext_vector_type(4)))  unsigned int v4u;
typedef __attribute__((ext_vector_type(4)))  unsigned int su4;  // SGPR quad
typedef __attribute__((ext_vector_type(8)))  unsigned int su8;  // SGPR octet

union Frag { v16bf bf; v4u u[2]; };

__device__ inline unsigned int pk_bf16(float a, float b) {
  unsigned int ua = __float_as_uint(a), ub = __float_as_uint(b);
  ua = (ua + 0x7FFFu + ((ua >> 16) & 1u)) >> 16;   // round-to-nearest-even
  ub = (ub + 0x7FFFu + ((ub >> 16) & 1u)) >> 16;
  return ua | (ub << 16);
}
__device__ inline unsigned short f2bf(float x) {
  unsigned int u = __float_as_uint(x);
  u += 0x7FFFu + ((u >> 16) & 1u);
  return (unsigned short)(u >> 16);
}

// Branch-free tanh: prefer gfx1250's hardware V_TANH_F32; otherwise an
// exp2/rcp form that saturates cleanly to +/-1 without branches.
__device__ inline float fast_tanh(float x) {
#if __has_builtin(__builtin_amdgcn_tanhf)
  return __builtin_amdgcn_tanhf(x);
#else
  float e = __builtin_amdgcn_exp2f(x * 2.8853900817779268f);  // exp(2x)
  return 1.0f - 2.0f * __builtin_amdgcn_rcpf(e + 1.0f);
#endif
}
__device__ inline float gelu_tanh(float x) {
  const float c = 0.7978845608028654f;
  float t = fast_tanh(c * (x + 0.044715f * x * x * x));
  return 0.5f * x * (1.0f + t);
}

// ---------------------------------------------------------------- router ----
__global__ __launch_bounds__(256) void moe_router(
    const float* __restrict__ X, const float* __restrict__ wg,
    const float* __restrict__ bg, float* __restrict__ gatebuf,
    float* __restrict__ auxacc) {
  __shared__ float red[8];  // [0..3] route counts, [4..7] prob sums
  const int lane = threadIdx.x & 31;
  const int t = blockIdx.x * 8 + (threadIdx.x >> 5);
  if (threadIdx.x < 8) red[threadIdx.x] = 0.0f;
  __syncthreads();

  float s0 = 0.f, s1 = 0.f, s2 = 0.f, s3 = 0.f;
  const float* xr = X + (size_t)t * H_DIM;
  for (int h = lane; h < H_DIM; h += 32) {
    float x = xr[h];
    float4 g = *(const float4*)(wg + h * N_EXP);  // one row == E=4 floats
    s0 += x * g.x; s1 += x * g.y; s2 += x * g.z; s3 += x * g.w;
  }
  for (int off = 16; off >= 1; off >>= 1) {
    s0 += __shfl_xor(s0, off, 32); s1 += __shfl_xor(s1, off, 32);
    s2 += __shfl_xor(s2, off, 32); s3 += __shfl_xor(s3, off, 32);
  }
  if (lane == 0) {
    float l[4] = { s0 + bg[0], s1 + bg[1], s2 + bg[2], s3 + bg[3] };
    float mx = fmaxf(fmaxf(l[0], l[1]), fmaxf(l[2], l[3]));
    float p[4], s = 0.f;
    for (int e = 0; e < 4; ++e) { p[e] = expf(l[e] - mx); s += p[e]; }
    float inv = 1.0f / s;
    for (int e = 0; e < 4; ++e) p[e] *= inv;
    int i0 = 0;
    for (int e = 1; e < 4; ++e) if (p[e] > p[i0]) i0 = e;
    int i1 = (i0 == 0) ? 1 : 0;
    for (int e = 0; e < 4; ++e) if (e != i0 && p[e] > p[i1]) i1 = e;
    float wsum = p[i0] + p[i1];
    float g4[4] = { 0.f, 0.f, 0.f, 0.f };
    g4[i0] = p[i0] / wsum;
    g4[i1] = p[i1] / wsum;
    *(float4*)(gatebuf + (size_t)t * 4) = make_float4(g4[0], g4[1], g4[2], g4[3]);
    atomicAdd(&red[i0], 1.0f);
    atomicAdd(&red[i1], 1.0f);
    for (int e = 0; e < 4; ++e) atomicAdd(&red[4 + e], p[e]);
  }
  __syncthreads();
  if (threadIdx.x < 8) atomicAdd(&auxacc[threadIdx.x], red[threadIdx.x]);
}

__global__ void moe_zero_aux(float* auxacc) {
  if (threadIdx.x < 8) auxacc[threadIdx.x] = 0.0f;
}

__global__ void moe_finalize(const float* __restrict__ auxacc,
                             float* __restrict__ out_aux) {
  float aux = 0.f;
  for (int e = 0; e < N_EXP; ++e) {
    float frac = auxacc[e] / (float)((size_t)T_TOK * TOP_K);
    float mp   = auxacc[4 + e] / (float)T_TOK;
    aux += frac * mp;
  }
  *out_aux = (float)N_EXP * aux;
}

// --------------------------------------------------------------- GEMM #1 ----
// h[t,f] = gelu(sum_k X[t,k]*W1[k,f] + b1[f]); h stored bf16 in workspace.
__global__ __launch_bounds__(256) void moe_gemm1(
    const float* __restrict__ A, const float* __restrict__ W,
    const float* __restrict__ bias, unsigned short* __restrict__ Hout) {
  __shared__ __align__(16) unsigned char smem[16384];
  unsigned char* As = smem;          // 128 rows x 32 bf16 (64B row stride)
  unsigned char* Bs = smem + 8192;   // [N=128][K=32] bf16 (transposed)

  const int tBase = blockIdx.y * 128, fBase = blockIdx.x * 128;
  const int tid = threadIdx.x, lane = tid & 31, wave = tid >> 5;
  const int waveM = wave >> 1, waveN = wave & 1;
  const int hi = (lane >> 4) & 1, cl = lane & 15;
  const int arow = tid >> 1, ahalf = tid & 1;       // A-tile fill role
  const int wk = tid >> 3, wn = (tid & 7) * 16;     // W-tile fill role

  v8f acc[2][4] = {};

  for (int k0 = 0; k0 < H_DIM; k0 += 32) {
    __syncthreads();
    {  // A tile: fp32 -> bf16, row major
      const float* src = A + (size_t)(tBase + arow) * H_DIM + k0 + ahalf * 16;
      float4 f0 = ((const float4*)src)[0], f1 = ((const float4*)src)[1];
      float4 f2 = ((const float4*)src)[2], f3 = ((const float4*)src)[3];
      v4u* dst = (v4u*)(As + arow * 64 + ahalf * 32);
      dst[0] = (v4u){ pk_bf16(f0.x, f0.y), pk_bf16(f0.z, f0.w),
                      pk_bf16(f1.x, f1.y), pk_bf16(f1.z, f1.w) };
      dst[1] = (v4u){ pk_bf16(f2.x, f2.y), pk_bf16(f2.z, f2.w),
                      pk_bf16(f3.x, f3.y), pk_bf16(f3.z, f3.w) };
    }
    {  // W tile: fp32 -> bf16, transposed into [N][K]
      const float* src = W + (size_t)(k0 + wk) * FF_DIM + fBase + wn;
      float4 f[4];
      f[0] = ((const float4*)src)[0]; f[1] = ((const float4*)src)[1];
      f[2] = ((const float4*)src)[2]; f[3] = ((const float4*)src)[3];
      const float* fs = (const float*)f;
      for (int m = 0; m < 16; ++m)
        *(unsigned short*)(Bs + (wn + m) * 64 + wk * 2) = f2bf(fs[m]);
    }
    __syncthreads();

    Frag a[2], b[4];
    for (int i = 0; i < 2; ++i) {
      int row = waveM * 32 + i * 16 + cl;
      const v4u* p = (const v4u*)(As + row * 64 + hi * 32);
      a[i].u[0] = p[0];   // K chunk [k0e, k0e+8)
      a[i].u[1] = p[1];   // K chunk [k0e+8, k0e+16)
    }
    for (int j = 0; j < 4; ++j) {
      int col = waveN * 64 + j * 16 + cl;
      const v4u* p = (const v4u*)(Bs + col * 64 + hi * 32);
      b[j].u[0] = p[0];
      b[j].u[1] = p[1];
    }
    for (int i = 0; i < 2; ++i)
      for (int j = 0; j < 4; ++j)
        acc[i][j] = __builtin_amdgcn_wmma_f32_16x16x32_bf16(
            false, a[i].bf, false, b[j].bf, (short)0, acc[i][j], false, false);
  }

  for (int i = 0; i < 2; ++i)
    for (int j = 0; j < 4; ++j) {
      int col = fBase + waveN * 64 + j * 16 + cl;
      float bb = bias[col];
      for (int v = 0; v < 8; ++v) {
        int row = tBase + waveM * 32 + i * 16 + hi * 8 + v;
        float val = gelu_tanh(acc[i][j][v] + bb);
        Hout[(size_t)row * FF_DIM + col] = f2bf(val);
      }
    }
}

// --------------------------------------------------------------- GEMM #2 ----
// out[t,h] (+)= gate[t,e] * (sum_f h[t,f]*W2[f,h] + b2[h])
// A-tile (raw bf16 copy): rows 0-63 via TDM tensor_load_to_lds (one D# per
// workgroup, issued by wave 0), rows 64-127 via per-lane async global->LDS.
__global__ __launch_bounds__(256) void moe_gemm2(
    const unsigned short* __restrict__ Hbuf, const float* __restrict__ W,
    const float* __restrict__ bias, const float* __restrict__ gate,
    int expert, float* __restrict__ Out, int accumulate) {
  __shared__ __align__(16) unsigned char smem[16384];
  unsigned char* As = smem;          // region starts at LDS offset 0
  unsigned char* Bs = smem + 8192;

  const int tBase = blockIdx.y * 128, hBase = blockIdx.x * 128;
  const int tid = threadIdx.x, lane = tid & 31, wave = tid >> 5;
  const int waveM = wave >> 1, waveN = wave & 1;
  const int hi = (lane >> 4) & 1, cl = lane & 15;
  const int arow = tid >> 1, ahalf = tid & 1;
  const int wk = tid >> 3, wn = (tid & 7) * 16;

  v8f acc[2][4] = {};

  // TDM descriptor group 1 (loop-invariant): data_size=2B, tensor [T,FF] bf16,
  // tile 32(x) x 64(y), dim0 stride = FF elements. (ISA 8.4 bitfields.)
  su8 g1;
  g1[0] = 0x00010000u;                                     // data_size=1 (2B)
  g1[1] = (unsigned)(FF_DIM & 0xFFFF) << 16;               // tensor_dim0 lo16
  g1[2] = (unsigned)(FF_DIM >> 16) |
          ((unsigned)(T_TOK & 0xFFFF) << 16);              // dim0 hi | dim1 lo
  g1[3] = (unsigned)(T_TOK >> 16) | (32u << 16);           // dim1 hi | tile_dim0
  g1[4] = 64u;                                             // tile_dim1=64 rows
  g1[5] = (unsigned)FF_DIM;                                // dim0_stride lo32
  g1[6] = 0u;                                              // stride hi, d1s lo
  g1[7] = 0u;                                              // d1 stride hi

  const unsigned ldsoff = (unsigned)(arow * 64 + ahalf * 32);

  for (int k0 = 0; k0 < FF_DIM; k0 += 32) {
    __syncthreads();
    if (tid < 32) {  // wave 0: TDM DMA for rows 0..63 of the A tile
      unsigned long long ga = (unsigned long long)(const void*)(
          Hbuf + (size_t)tBase * FF_DIM + k0);
      su4 g0;
      g0[0] = 1u;                                          // count=1 (valid)
      g0[1] = 0u;                                          // lds_addr = As+0
      g0[2] = (unsigned)ga;                                // global_addr lo32
      g0[3] = (unsigned)((ga >> 32) & 0x1FFFFFFu) | (2u << 30);  // hi | type=2
      asm volatile("tensor_load_to_lds %0, %1"
                   :: "s"(g0), "s"(g1) : "memory");
    }
    if (tid >= 128) {  // waves 4-7: async DMA for rows 64..127
      unsigned long long ga = (unsigned long long)(const void*)(
          Hbuf + (size_t)(tBase + arow) * FF_DIM + k0 + ahalf * 16);
      asm volatile(
          "global_load_async_to_lds_b128 %0, %1, off\n\t"
          "global_load_async_to_lds_b128 %0, %1, off offset:16"
          :: "v"(ldsoff), "v"(ga) : "memory");
    }
    {  // W2 tile: fp32 -> bf16, transposed
      const float* src = W + (size_t)(k0 + wk) * H_DIM + hBase + wn;
      float4 f[4];
      f[0] = ((const float4*)src)[0]; f[1] = ((const float4*)src)[1];
      f[2] = ((const float4*)src)[2]; f[3] = ((const float4*)src)[3];
      const float* fs = (const float*)f;
      for (int m = 0; m < 16; ++m)
        *(unsigned short*)(Bs + (wn + m) * 64 + wk * 2) = f2bf(fs[m]);
    }
    // Drain this wave's outstanding DMA (counts are 0 where not issued),
    // then publish LDS to the workgroup.
    asm volatile("s_wait_asynccnt 0x0\n\ts_wait_tensorcnt 0x0" ::: "memory");
    __syncthreads();

    Frag a[2], b[4];
    for (int i = 0; i < 2; ++i) {
      int row = waveM * 32 + i * 16 + cl;
      const v4u* p = (const v4u*)(As + row * 64 + hi * 32);
      a[i].u[0] = p[0];
      a[i].u[1] = p[1];
    }
    for (int j = 0; j < 4; ++j) {
      int col = waveN * 64 + j * 16 + cl;
      const v4u* p = (const v4u*)(Bs + col * 64 + hi * 32);
      b[j].u[0] = p[0];
      b[j].u[1] = p[1];
    }
    for (int i = 0; i < 2; ++i)
      for (int j = 0; j < 4; ++j)
        acc[i][j] = __builtin_amdgcn_wmma_f32_16x16x32_bf16(
            false, a[i].bf, false, b[j].bf, (short)0, acc[i][j], false, false);
  }

  for (int i = 0; i < 2; ++i) {
    float gv[8];
    for (int v = 0; v < 8; ++v) {
      int row = tBase + waveM * 32 + i * 16 + hi * 8 + v;
      gv[v] = gate[(size_t)row * N_EXP + expert];
    }
    for (int j = 0; j < 4; ++j) {
      int col = hBase + waveN * 64 + j * 16 + cl;
      float bb = bias[col];
      for (int v = 0; v < 8; ++v) {
        int row = tBase + waveM * 32 + i * 16 + hi * 8 + v;
        float val = gv[v] * (acc[i][j][v] + bb);
        size_t o = (size_t)row * H_DIM + col;
        Out[o] = accumulate ? (Out[o] + val) : val;
      }
    }
  }
}

// ---------------------------------------------------------------- launch ----
extern "C" void kernel_launch(void* const* d_in, const int* in_sizes, int n_in,
                              void* d_out, int out_size, void* d_ws,
                              size_t ws_size, hipStream_t stream) {
  const float* X  = (const float*)d_in[0];  // [B,S,H]
  const float* wg = (const float*)d_in[1];  // [H,E]
  const float* bg = (const float*)d_in[2];  // [E]
  const float* w1 = (const float*)d_in[3];  // [E,H,FF]
  const float* b1 = (const float*)d_in[4];  // [E,FF]
  const float* w2 = (const float*)d_in[5];  // [E,FF,H]
  const float* b2 = (const float*)d_in[6];  // [E,H]
  float* out = (float*)d_out;               // [B,S,H] then aux_loss scalar

  // Workspace layout: [0,32) aux accumulators; [256,...) gate table [T,4];
  // then bf16 h buffer [T,FF].
  float* auxacc  = (float*)d_ws;
  float* gatebuf = (float*)((char*)d_ws + 256);
  unsigned short* hbuf =
      (unsigned short*)((char*)d_ws + 256 + (size_t)T_TOK * N_EXP * 4);

  moe_zero_aux<<<1, 32, 0, stream>>>(auxacc);
  moe_router<<<T_TOK / 8, 256, 0, stream>>>(X, wg, bg, gatebuf, auxacc);

  for (int e = 0; e < N_EXP; ++e) {
    moe_gemm1<<<dim3(FF_DIM / 128, T_TOK / 128), 256, 0, stream>>>(
        X, w1 + (size_t)e * H_DIM * FF_DIM, b1 + (size_t)e * FF_DIM, hbuf);
    moe_gemm2<<<dim3(H_DIM / 128, T_TOK / 128), 256, 0, stream>>>(
        hbuf, w2 + (size_t)e * FF_DIM * H_DIM, b2 + (size_t)e * H_DIM, gatebuf,
        e, out, e > 0);
  }

  moe_finalize<<<1, 1, 0, stream>>>(auxacc, out + (size_t)T_TOK * H_DIM);
}
